// TransLayer_10926396801688
// MI455X (gfx1250) — matrix operand order repaired and checked by reading, not compiled, not run on hardware
//
#include <hip/hip_runtime.h>

// ---------------------------------------------------------------------------
// Types
// ---------------------------------------------------------------------------
typedef unsigned short u16;
typedef unsigned int   u32;
typedef __attribute__((ext_vector_type(8)))  u32   u8v;
typedef __attribute__((ext_vector_type(16))) __bf16 v16bf;
typedef __attribute__((ext_vector_type(8)))  float v8f;
typedef __attribute__((ext_vector_type(2)))  float v2f;

// Problem constants (fixed by setup_inputs)
#define BB   4
#define NN   4096
#define DD   512
#define HH   8
#define DHH  64
#define MM   256      // landmarks
#define BH   32       // BB*HH
#define NTOK 16384    // BB*NN

__device__ __forceinline__ u16 f2bf(float f) {
    u32 u = __builtin_bit_cast(u32, f);
    u32 r = (u + 0x7FFFu + ((u >> 16) & 1u)) >> 16;
    return (u16)r;
}
__device__ __forceinline__ float bf2f(u16 h) {
    u32 u = ((u32)h) << 16;
    return __builtin_bit_cast(float, u);
}

// ---------------------------------------------------------------------------
// Block reductions (blockDim.x == 256)
// ---------------------------------------------------------------------------
__device__ __forceinline__ float block_sum(float v, float* sh) {
    int t = threadIdx.x;
    sh[t] = v; __syncthreads();
    for (int s = 128; s > 0; s >>= 1) {
        if (t < s) sh[t] += sh[t + s];
        __syncthreads();
    }
    float r = sh[0]; __syncthreads();
    return r;
}
__device__ __forceinline__ float block_max(float v, float* sh) {
    int t = threadIdx.x;
    sh[t] = v; __syncthreads();
    for (int s = 128; s > 0; s >>= 1) {
        if (t < s) sh[t] = fmaxf(sh[t], sh[t + s]);
        __syncthreads();
    }
    float r = sh[0]; __syncthreads();
    return r;
}

// ---------------------------------------------------------------------------
// Tiled bf16 GEMM (NT): C[m,n] = act( A[m,:] . B[n,:] + bias[n] ) + resid
//   A: [Md,K] bf16 row-major    B: [Nd,K] bf16 row-major (op(B)=B^T)
//   Block tile 128x128, 8 waves; per-wave 64x32 (4x2 wmma tiles).
//   Tiles staged in LDS with global_load_async_to_lds_b128 (ASYNCcnt path),
//   double-buffered so the async DMA overlaps the WMMA stream.
//   Requires: Md%128==0, Nd%128==0, K%32==0.
// ---------------------------------------------------------------------------
__global__ __launch_bounds__(256)
void gemm_bf16_tiled(const u16* __restrict__ A, const u16* __restrict__ B,
                     float* __restrict__ Cf, u16* __restrict__ Cb,
                     const float* __restrict__ bias, const float* __restrict__ resid,
                     int Md, int Nd, int K,
                     long long sA, long long sB, long long sC, int gelu)
{
    __shared__ u16 As[2][128 * 32];
    __shared__ u16 Bs[2][128 * 32];

    const int tid  = threadIdx.x;
    const int lane = tid & 31;
    const int w    = tid >> 5;       // wave 0..7
    const int wy   = w >> 2;         // 0..1  -> 64 rows of M
    const int wx   = w & 3;          // 0..3  -> 32 cols of N
    const int r16  = lane & 15;
    const int half = lane >> 4;

    const int m0 = blockIdx.y * 128;
    const int n0 = blockIdx.x * 128;
    const int bz = blockIdx.z;

    const u16* Ab = A + (size_t)bz * sA;
    const u16* Bb = B + (size_t)bz * sB;

    // Loader mapping: 2 threads per tile row, 32B (2 x b128) each.
    const int lrow = tid >> 1;          // 0..127
    const int lseg = (tid & 1) * 16;    // element offset 0 / 16

    v8f acc[4][2];
#pragma unroll
    for (int i = 0; i < 4; ++i)
#pragma unroll
        for (int j = 0; j < 2; ++j)
            acc[i][j] = (v8f){0.f, 0.f, 0.f, 0.f, 0.f, 0.f, 0.f, 0.f};

    const int nk = K / 32;

    // Issue async copy of one 128x32 A tile + 128x32 B tile into buffer `buf`
    auto issue_tile = [&](int buf, int k0) {
        const u16* ga = Ab + (size_t)(m0 + lrow) * K + k0 + lseg;
        const u16* gb = Bb + (size_t)(n0 + lrow) * K + k0 + lseg;
        unsigned la = (unsigned)(size_t)&As[buf][lrow * 32 + lseg];
        unsigned lb = (unsigned)(size_t)&Bs[buf][lrow * 32 + lseg];
        asm volatile(
            "global_load_async_to_lds_b128 %0, %1, off\n\t"
            "global_load_async_to_lds_b128 %0, %1, off offset:16\n\t"
            "global_load_async_to_lds_b128 %2, %3, off\n\t"
            "global_load_async_to_lds_b128 %2, %3, off offset:16"
            :: "v"(la), "v"(ga), "v"(lb), "v"(gb) : "memory");
    };

    issue_tile(0, 0);

    for (int kt = 0; kt < nk; ++kt) {
        const int cur = kt & 1;
        asm volatile("s_wait_asynccnt 0x0" ::: "memory");
        __syncthreads();                    // tile `cur` visible to all waves
        if (kt + 1 < nk) issue_tile(cur ^ 1, (kt + 1) * 32);

        // B fragments (2 x 16 cols)
        v16bf bf[2];
#pragma unroll
        for (int j = 0; j < 2; ++j) {
            const u16* pb = &Bs[cur][(wx * 32 + j * 16 + r16) * 32 + half * 8];
            uint4 b0 = *reinterpret_cast<const uint4*>(pb);
            uint4 b1 = *reinterpret_cast<const uint4*>(pb + 16);
            u8v bu = {b0.x, b0.y, b0.z, b0.w, b1.x, b1.y, b1.z, b1.w};
            bf[j] = __builtin_bit_cast(v16bf, bu);
        }
        // A fragments (4 x 16 rows), each feeds 2 wmmas
#pragma unroll
        for (int i = 0; i < 4; ++i) {
            const u16* pa = &As[cur][(wy * 64 + i * 16 + r16) * 32 + half * 8];
            uint4 a0 = *reinterpret_cast<const uint4*>(pa);
            uint4 a1 = *reinterpret_cast<const uint4*>(pa + 16);
            u8v au = {a0.x, a0.y, a0.z, a0.w, a1.x, a1.y, a1.z, a1.w};
            v16bf av = __builtin_bit_cast(v16bf, au);
#pragma unroll
            for (int j = 0; j < 2; ++j)
                acc[i][j] = __builtin_amdgcn_wmma_f32_16x16x32_bf16(
                    false, av, false, bf[j], (short)0, acc[i][j], false, false);
        }
        __syncthreads();                    // everyone done reading `cur`
    }

    // Epilogue
#pragma unroll
    for (int i = 0; i < 4; ++i) {
#pragma unroll
        for (int j = 0; j < 2; ++j) {
            const int nn = n0 + wx * 32 + j * 16 + r16;
            const float bi = bias ? bias[nn] : 0.f;
#pragma unroll
            for (int r = 0; r < 8; ++r) {
                int mm = m0 + wy * 64 + i * 16 + half * 8 + r;
                float v = acc[i][j][r] + bi;
                if (gelu) v = 0.5f * v * (1.f + erff(v * 0.70710678118654752f));
                size_t idx = (size_t)bz * sC + (size_t)mm * Nd + nn;
                if (resid) v += resid[idx];
                if (Cf) Cf[idx] = v;
                if (Cb) Cb[idx] = f2bf(v);
            }
        }
    }
}

// ---------------------------------------------------------------------------
// Simple bf16 GEMM (NT), one wave per 16x16 tile — used for Nd==64 cases.
// ---------------------------------------------------------------------------
__global__ __launch_bounds__(128)
void gemm_bf16_nt(const u16* __restrict__ A, const u16* __restrict__ B,
                  float* __restrict__ Cf, u16* __restrict__ Cb,
                  const float* __restrict__ bias, const float* __restrict__ resid,
                  int Md, int Nd, int K,
                  long long sA, long long sB, long long sC, int gelu)
{
    const int lane = threadIdx.x;
    const int r16  = lane & 15;
    const int half = lane >> 4;
    const int n0   = (blockIdx.x * 4 + threadIdx.y) * 16;
    const int m0   = blockIdx.y * 16;
    const int bz   = blockIdx.z;

    const u16* Ab = A + (size_t)bz * sA;
    const u16* Bb = B + (size_t)bz * sB;

    v8f acc = {0.f, 0.f, 0.f, 0.f, 0.f, 0.f, 0.f, 0.f};

    for (int k0 = 0; k0 < K; k0 += 32) {
        const u16* pa = Ab + (size_t)(m0 + r16) * K + k0 + half * 8;
        uint4 a0 = *reinterpret_cast<const uint4*>(pa);
        uint4 a1 = *reinterpret_cast<const uint4*>(pa + 16);
        const u16* pb = Bb + (size_t)(n0 + r16) * K + k0 + half * 8;
        uint4 b0 = *reinterpret_cast<const uint4*>(pb);
        uint4 b1 = *reinterpret_cast<const uint4*>(pb + 16);

        u8v au = {a0.x, a0.y, a0.z, a0.w, a1.x, a1.y, a1.z, a1.w};
        u8v bu = {b0.x, b0.y, b0.z, b0.w, b1.x, b1.y, b1.z, b1.w};
        v16bf av = __builtin_bit_cast(v16bf, au);
        v16bf bv = __builtin_bit_cast(v16bf, bu);

        acc = __builtin_amdgcn_wmma_f32_16x16x32_bf16(
            false, av, false, bv, (short)0, acc, false, false);
    }

    const int nn = n0 + r16;
    const float bi = bias ? bias[nn] : 0.f;
#pragma unroll
    for (int r = 0; r < 8; ++r) {
        int mm = m0 + half * 8 + r;
        float v = acc[r] + bi;
        if (gelu) v = 0.5f * v * (1.f + erff(v * 0.70710678118654752f));
        size_t idx = (size_t)bz * sC + (size_t)mm * Nd + nn;
        if (resid) v += resid[idx];
        if (Cf) Cf[idx] = v;
        if (Cb) Cb[idx] = f2bf(v);
    }
}

// ---------------------------------------------------------------------------
// f32 GEMM for the pinv iteration (256x256x256, batch 32) via wmma 16x16x4 f32.
//   outP = scaleP * (A@B)   (optional)     outI = alphaI*I - (A@B)   (optional)
// ---------------------------------------------------------------------------
__global__ __launch_bounds__(128)
void gemm_f32_pinv(const float* __restrict__ A, const float* __restrict__ B,
                   float* __restrict__ outP, float* __restrict__ outI,
                   float scaleP, float alphaI)
{
    const int lane = threadIdx.x;
    const int r16  = lane & 15;
    const int half = lane >> 4;
    const int n0   = (blockIdx.x * 4 + threadIdx.y) * 16;
    const int m0   = blockIdx.y * 16;
    const size_t base = (size_t)blockIdx.z * 65536;

    const float* Ab = A + base;
    const float* Bb = B + base;

    v8f acc = {0.f, 0.f, 0.f, 0.f, 0.f, 0.f, 0.f, 0.f};

    for (int k0 = 0; k0 < 256; k0 += 4) {
        int kb = k0 + 2 * half;
        float2 af = *reinterpret_cast<const float2*>(Ab + (size_t)(m0 + r16) * 256 + kb);
        v2f a = {af.x, af.y};
        v2f b = {Bb[(size_t)kb * 256 + n0 + r16],
                 Bb[(size_t)(kb + 1) * 256 + n0 + r16]};
        acc = __builtin_amdgcn_wmma_f32_16x16x4_f32(
            false, a, false, b, (short)0, acc, false, false);
    }

    const int nn = n0 + r16;
#pragma unroll
    for (int r = 0; r < 8; ++r) {
        int mm = m0 + half * 8 + r;
        size_t idx = base + (size_t)mm * 256 + nn;
        float v = acc[r];
        if (outP) outP[idx] = scaleP * v;
        if (outI) outI[idx] = (mm == nn ? alphaI : 0.f) - v;
    }
}

// ---------------------------------------------------------------------------
// LayerNorm over D=512, one block (256 thr) per token row, bf16 output
// ---------------------------------------------------------------------------
__global__ __launch_bounds__(256)
void ln_kernel(const float* __restrict__ x, const float* __restrict__ g,
               const float* __restrict__ b, u16* __restrict__ out)
{
    __shared__ float sh[256];
    const size_t row = blockIdx.x;
    const float* xr = x + row * DD;
    float s = 0.f;
    for (int i = threadIdx.x; i < DD; i += 256) s += xr[i];
    float mean = block_sum(s, sh) * (1.f / DD);
    float s2 = 0.f;
    for (int i = threadIdx.x; i < DD; i += 256) { float d = xr[i] - mean; s2 += d * d; }
    float rstd = rsqrtf(block_sum(s2, sh) * (1.f / DD) + 1e-5f);
    for (int i = threadIdx.x; i < DD; i += 256)
        out[row * DD + i] = f2bf((xr[i] - mean) * rstd * g[i] + b[i]);
}

// ---------------------------------------------------------------------------
// Row softmax, len arbitrary, outputs f32 and/or bf16
// ---------------------------------------------------------------------------
__global__ __launch_bounds__(256)
void softmax_rows(const float* __restrict__ in, float* __restrict__ outF,
                  u16* __restrict__ outB, int len)
{
    __shared__ float sh[256];
    const size_t row = blockIdx.x;
    const float* p = in + row * (size_t)len;
    float mx = -3.4e38f;
    for (int i = threadIdx.x; i < len; i += 256) mx = fmaxf(mx, p[i]);
    mx = block_max(mx, sh);
    float sum = 0.f;
    for (int i = threadIdx.x; i < len; i += 256) sum += __expf(p[i] - mx);
    float inv = 1.f / block_sum(sum, sh);
    for (int i = threadIdx.x; i < len; i += 256) {
        float e = __expf(p[i] - mx) * inv;
        size_t idx = row * (size_t)len + i;
        if (outF) outF[idx] = e;
        if (outB) outB[idx] = f2bf(e);
    }
}

// ---------------------------------------------------------------------------
// Split qkv f32 [tok,1536] -> per-head bf16 q (scaled), k, v, and v^T
// ---------------------------------------------------------------------------
__global__ void split_qkv(const float* __restrict__ qkv, u16* __restrict__ qb,
                          u16* __restrict__ kb, u16* __restrict__ vb,
                          u16* __restrict__ vt)
{
    size_t idx = (size_t)blockIdx.x * 256 + threadIdx.x;
    if (idx >= (size_t)NTOK * DD) return;
    int c  = idx & 63;
    int h  = (idx >> 6) & 7;
    size_t t = idx >> 9;
    int b  = (int)(t >> 12);
    int i  = (int)(t & 4095);
    const float* base = qkv + t * 1536 + h * 64 + c;
    float qv = base[0] * 0.125f;   // DH^-0.5
    float kv = base[512];
    float vv = base[1024];
    size_t bh = (size_t)b * HH + h;
    size_t o  = (bh * NN + i) * DHH + c;
    qb[o] = f2bf(qv);
    kb[o] = f2bf(kv);
    vb[o] = f2bf(vv);
    vt[(bh * DHH + c) * NN + i] = f2bf(vv);
}

// ---------------------------------------------------------------------------
// Landmark means over groups of 16 tokens (from f32 qkv for precision)
// ---------------------------------------------------------------------------
__global__ void landmarks(const float* __restrict__ qkv, u16* __restrict__ ql,
                          u16* __restrict__ kl)
{
    size_t idx = (size_t)blockIdx.x * 256 + threadIdx.x;
    if (idx >= (size_t)BH * MM * DHH) return;
    int c  = idx & 63;
    int mi = (int)((idx >> 6) & 255);
    int bh = (int)(idx >> 14);
    int b  = bh >> 3, h = bh & 7;
    float sq = 0.f, sk = 0.f;
    for (int j = 0; j < 16; ++j) {
        size_t t = (size_t)b * NN + mi * 16 + j;
        const float* base = qkv + t * 1536 + h * 64 + c;
        sq += base[0];
        sk += base[512];
    }
    size_t o = ((size_t)bh * MM + mi) * DHH + c;
    ql[o] = f2bf(sq * (0.125f / 16.f));
    kl[o] = f2bf(sk * (1.f / 16.f));
}

// ---------------------------------------------------------------------------
// Batched f32 -> bf16 transpose: out[b][c][r] = in[b][r][c]
// ---------------------------------------------------------------------------
__global__ void transpose_to_bf16(const float* __restrict__ in, u16* __restrict__ out,
                                  int rows, int cols)
{
    size_t per = (size_t)rows * cols;
    size_t idx = (size_t)blockIdx.x * 256 + threadIdx.x;
    if (idx >= per) return;
    size_t b = blockIdx.y;
    int r = (int)(idx / cols);
    int c = (int)(idx % cols);
    out[b * per + (size_t)c * rows + r] = f2bf(in[b * per + idx]);
}

// ---------------------------------------------------------------------------
// pinv: global max of row-sums / col-sums of |x| via float-bits atomicMax
// ---------------------------------------------------------------------------
__global__ void init_scal(float* scal) { if (threadIdx.x < 2) scal[threadIdx.x] = 0.f; }

__global__ __launch_bounds__(256)
void pinv_norms(const float* __restrict__ x, float* __restrict__ scal)
{
    __shared__ float sh[256];
    const int bh = blockIdx.x;
    const int j  = threadIdx.x;
    const float* base = x + (size_t)bh * 65536;
    float rs = 0.f, cs = 0.f;
    for (int c = 0; c < 256; ++c) {
        rs += fabsf(base[(size_t)j * 256 + c]);   // row sum  (axis=-1)
        cs += fabsf(base[(size_t)c * 256 + j]);   // col sum  (axis=-2)
    }
    rs = block_max(rs, sh);
    cs = block_max(cs, sh);
    if (threadIdx.x == 0) {
        atomicMax((u32*)&scal[0], __builtin_bit_cast(u32, rs));  // col = max row-sums
        atomicMax((u32*)&scal[1], __builtin_bit_cast(u32, cs));  // row = max col-sums
    }
}

__global__ void pinv_init_z(const float* __restrict__ x, float* __restrict__ z,
                            const float* __restrict__ scal)
{
    size_t idx = (size_t)blockIdx.x * 256 + threadIdx.x;
    if (idx >= (size_t)BH * 65536) return;
    size_t bh = idx >> 16;
    int i = (int)((idx >> 8) & 255);
    int j = (int)(idx & 255);
    float inv = 1.f / (scal[0] * scal[1]);
    z[idx] = x[(bh << 16) + ((size_t)j << 8) + i] * inv;   // x^T / (col*row)
}

// ---------------------------------------------------------------------------
// Depthwise conv (kernel 33, pad 16) over sequence + merge heads -> bf16
// ---------------------------------------------------------------------------
__global__ void conv_merge(const float* __restrict__ oh, const u16* __restrict__ vb,
                           const float* __restrict__ wconv, u16* __restrict__ merged)
{
    size_t idx = (size_t)blockIdx.x * 256 + threadIdx.x;
    if (idx >= (size_t)BH * NN * DHH) return;
    int c  = idx & 63;
    int i  = (int)((idx >> 6) & 4095);
    int bh = (int)(idx >> 18);
    int h  = bh & 7, b = bh >> 3;
    float acc = oh[idx];
    const u16* vrow = vb + ((size_t)bh * NN) * DHH + c;
    for (int t = 0; t < 33; ++t) {
        int ii = i + t - 16;
        if (ii >= 0 && ii < NN)
            acc += bf2f(vrow[(size_t)ii * DHH]) * wconv[h * 33 + t];
    }
    merged[((size_t)b * NN + i) * DD + h * 64 + c] = f2bf(acc);
}

// ---------------------------------------------------------------------------
// Host: full layer pipeline
// ---------------------------------------------------------------------------
extern "C" void kernel_launch(void* const* d_in, const int* in_sizes, int n_in,
                              void* d_out, int out_size, void* d_ws, size_t ws_size,
                              hipStream_t stream)
{
    (void)in_sizes; (void)n_in; (void)out_size; (void)ws_size;
    const float* x        = (const float*)d_in[0];
    const float* ln1_g    = (const float*)d_in[1];
    const float* ln1_b    = (const float*)d_in[2];
    const float* w_qkv    = (const float*)d_in[3];
    const float* res_conv = (const float*)d_in[4];
    const float* w_out    = (const float*)d_in[5];
    const float* b_out    = (const float*)d_in[6];
    const float* ln2_g    = (const float*)d_in[7];
    const float* ln2_b    = (const float*)d_in[8];
    const float* w_ff1    = (const float*)d_in[9];
    const float* b_ff1    = (const float*)d_in[10];
    const float* w_ff2    = (const float*)d_in[11];
    const float* b_ff2    = (const float*)d_in[12];
    float* out = (float*)d_out;

    char* ws = (char*)d_ws;
    size_t off = 0;
    auto alloc = [&](size_t bytes) -> char* {
        char* p = ws + off;
        off = (off + bytes + 255) & ~(size_t)255;
        return p;
    };

    u16*   wqkv_t = (u16*)alloc((size_t)1536 * 512 * 2);
    u16*   wout_t = (u16*)alloc((size_t)512 * 512 * 2);
    u16*   wff1_t = (u16*)alloc((size_t)2048 * 512 * 2);
    u16*   wff2_t = (u16*)alloc((size_t)512 * 2048 * 2);
    u16*   xn     = (u16*)alloc((size_t)NTOK * DD * 2);
    float* qkv    = (float*)alloc((size_t)NTOK * 1536 * 4);           // reused -> attn1b
    u16*   qb     = (u16*)alloc((size_t)BH * NN * DHH * 2);
    u16*   kb     = (u16*)alloc((size_t)BH * NN * DHH * 2);
    u16*   vb     = (u16*)alloc((size_t)BH * NN * DHH * 2);
    u16*   vt     = (u16*)alloc((size_t)BH * NN * DHH * 2);
    u16*   ql     = (u16*)alloc((size_t)BH * MM * DHH * 2);
    u16*   kl     = (u16*)alloc((size_t)BH * MM * DHH * 2);
    float* simbig = (float*)alloc((size_t)BH * NN * MM * 4);          // sim1 f32 -> sim3 f32 -> (w1b/h2b alias)
    u16*   attn3b = (u16*)alloc((size_t)BH * MM * NN * 2);
    float* sim2   = (float*)alloc((size_t)BH * MM * MM * 4);
    float* attn2  = (float*)alloc((size_t)BH * MM * MM * 4);
    float* zA     = (float*)alloc((size_t)BH * MM * MM * 4);
    float* zB     = (float*)alloc((size_t)BH * MM * MM * 4);
    float* xz     = (float*)alloc((size_t)BH * MM * MM * 4);
    float* tb     = (float*)alloc((size_t)BH * MM * MM * 4);
    float* t2b    = (float*)alloc((size_t)BH * MM * MM * 4);
    u16*   zt     = (u16*)alloc((size_t)BH * MM * MM * 2);
    float* kvf    = (float*)alloc((size_t)BH * MM * DHH * 4);
    u16*   kvt    = (u16*)alloc((size_t)BH * DHH * MM * 2);
    float* ohb    = (float*)alloc((size_t)BH * NN * DHH * 4);
    u16*   merged = (u16*)alloc((size_t)NTOK * DD * 2);
    float* yb     = (float*)alloc((size_t)NTOK * DD * 4);
    u16*   hb     = (u16*)alloc((size_t)NTOK * DD * 2);
    float* scal   = (float*)alloc(256);

    // Aliases over dead regions
    u16* attn1b = (u16*)qkv;                                           // 64MB in 96MB
    u16* w1b    = (u16*)((char*)simbig + (size_t)BH * NN * MM * 2);    // 2nd half of simbig
    u16* h2b    = (u16*)simbig;                                        // 1st half of simbig

    const dim3 blk(32, 4);
    const dim3 tblk(256);

    // Weight convert+transpose to bf16 [N,K]
    transpose_to_bf16<<<dim3((512 * 1536 + 255) / 256, 1), 256, 0, stream>>>(w_qkv, wqkv_t, 512, 1536);
    transpose_to_bf16<<<dim3((512 * 512 + 255) / 256, 1), 256, 0, stream>>>(w_out, wout_t, 512, 512);
    transpose_to_bf16<<<dim3((512 * 2048 + 255) / 256, 1), 256, 0, stream>>>(w_ff1, wff1_t, 512, 2048);
    transpose_to_bf16<<<dim3((2048 * 512 + 255) / 256, 1), 256, 0, stream>>>(w_ff2, wff2_t, 2048, 512);

    // LN1 -> bf16
    ln_kernel<<<NTOK, 256, 0, stream>>>(x, ln1_g, ln1_b, xn);

    // QKV GEMM: [16384,512] @ [512,1536]
    gemm_bf16_tiled<<<dim3(1536 / 128, NTOK / 128, 1), tblk, 0, stream>>>(
        xn, wqkv_t, qkv, nullptr, nullptr, nullptr, NTOK, 1536, 512, 0, 0, 0, 0);

    split_qkv<<<(NTOK * DD + 255) / 256, 256, 0, stream>>>(qkv, qb, kb, vb, vt);
    landmarks<<<(BH * MM * DHH + 255) / 256, 256, 0, stream>>>(qkv, ql, kl);

    // sim1 = q @ k_land^T : [4096,64]x[256,64]^T, batch 32
    gemm_bf16_tiled<<<dim3(MM / 128, NN / 128, BH), tblk, 0, stream>>>(
        qb, kl, simbig, nullptr, nullptr, nullptr, NN, MM, DHH,
        (long long)NN * DHH, (long long)MM * DHH, (long long)NN * MM, 0);
    softmax_rows<<<BH * NN, 256, 0, stream>>>(simbig, nullptr, attn1b, MM);

    // sim2 = q_land @ k_land^T : [256,64]x[256,64]^T
    gemm_bf16_tiled<<<dim3(MM / 128, MM / 128, BH), tblk, 0, stream>>>(
        ql, kl, sim2, nullptr, nullptr, nullptr, MM, MM, DHH,
        (long long)MM * DHH, (long long)MM * DHH, (long long)MM * MM, 0);
    softmax_rows<<<BH * MM, 256, 0, stream>>>(sim2, attn2, nullptr, MM);

    // sim3 = q_land @ k^T : [256,64]x[4096,64]^T
    gemm_bf16_tiled<<<dim3(NN / 128, MM / 128, BH), tblk, 0, stream>>>(
        ql, kb, simbig, nullptr, nullptr, nullptr, MM, NN, DHH,
        (long long)MM * DHH, (long long)NN * DHH, (long long)MM * NN, 0);
    softmax_rows<<<BH * MM, 256, 0, stream>>>(simbig, nullptr, attn3b, NN);

    // Moore-Penrose pinv of attn2 (fp32 WMMA)
    init_scal<<<1, 32, 0, stream>>>(scal);
    pinv_norms<<<BH, 256, 0, stream>>>(attn2, scal);
    pinv_init_z<<<(BH * 65536 + 255) / 256, 256, 0, stream>>>(attn2, zA, scal);

    float* zc = zA; float* zn = zB;
    const dim3 pg(MM / 64, MM / 16, BH);
    for (int it = 0; it < 6; ++it) {
        gemm_f32_pinv<<<pg, blk, 0, stream>>>(attn2, zc, xz, tb, 1.f, 7.f);      // xz, t=7I-xz
        gemm_f32_pinv<<<pg, blk, 0, stream>>>(xz, tb, nullptr, t2b, 0.f, 15.f);  // t2=15I-xz@t
        gemm_f32_pinv<<<pg, blk, 0, stream>>>(xz, t2b, nullptr, tb, 0.f, 13.f);  // t=13I-xz@t2
        gemm_f32_pinv<<<pg, blk, 0, stream>>>(zc, tb, zn, nullptr, 0.25f, 0.f);  // z=0.25*z@t
        float* tmp = zc; zc = zn; zn = tmp;
    }
    transpose_to_bf16<<<dim3((MM * MM + 255) / 256, BH), 256, 0, stream>>>(zc, zt, MM, MM);

    // kv = attn3 @ v : [256,4096] x [64,4096]^T (vt)  (Nd=64 -> simple kernel)
    gemm_bf16_nt<<<dim3(DHH / 64, MM / 16, BH), blk, 0, stream>>>(
        attn3b, vt, kvf, nullptr, nullptr, nullptr, MM, DHH, NN,
        (long long)MM * NN, (long long)DHH * NN, (long long)MM * DHH, 0);
    transpose_to_bf16<<<dim3((MM * DHH + 255) / 256, BH), 256, 0, stream>>>(kvf, kvt, MM, DHH);

    // w1 = attn1 @ attn2_inv : [4096,256] x [256,256]^T (zt) -> bf16 direct
    gemm_bf16_tiled<<<dim3(MM / 128, NN / 128, BH), tblk, 0, stream>>>(
        attn1b, zt, nullptr, w1b, nullptr, nullptr, NN, MM, MM,
        (long long)NN * MM, (long long)MM * MM, (long long)NN * MM, 0);

    // out_head = w1 @ kv : [4096,256] x [64,256]^T (kvt)  (Nd=64 -> simple kernel)
    gemm_bf16_nt<<<dim3(DHH / 64, NN / 16, BH), blk, 0, stream>>>(
        w1b, kvt, ohb, nullptr, nullptr, nullptr, NN, DHH, MM,
        (long long)NN * MM, (long long)DHH * MM, (long long)NN * DHH, 0);

    // + depthwise conv residual, merge heads -> bf16 [16384,512]
    conv_merge<<<(BH * NN * DHH + 255) / 256, 256, 0, stream>>>(ohb, vb, res_conv, merged);

    // out-proj + bias + residual x -> y f32
    gemm_bf16_tiled<<<dim3(DD / 128, NTOK / 128, 1), tblk, 0, stream>>>(
        merged, wout_t, yb, nullptr, b_out, x, NTOK, DD, DD, 0, 0, 0, 0);

    // LN2 -> bf16
    ln_kernel<<<NTOK, 256, 0, stream>>>(yb, ln2_g, ln2_b, hb);

    // FF1 + bias + exact GELU -> bf16
    gemm_bf16_tiled<<<dim3(2048 / 128, NTOK / 128, 1), tblk, 0, stream>>>(
        hb, wff1_t, nullptr, h2b, b_ff1, nullptr, NTOK, 2048, DD, 0, 0, 0, 1);

    // FF2 + bias + residual y -> d_out f32
    gemm_bf16_tiled<<<dim3(DD / 128, NTOK / 128, 1), tblk, 0, stream>>>(
        h2b, wff2_t, out, nullptr, b_ff2, yb, NTOK, DD, 2048, 0, 0, 0, 0);
}
